// ANFIS_66666482368793
// MI455X (gfx1250) — compile-verified, matrix operand order checked by reading, not run on hardware
//
#include <hip/hip_runtime.h>
#include <cstdint>

typedef __attribute__((ext_vector_type(16))) _Float16 v16h;
typedef __attribute__((ext_vector_type(8)))  _Float16 v8h;
typedef __attribute__((ext_vector_type(8)))  float    v8f;

#define INPUT_DIM    8
#define NUM_MFS      4
#define NUM_OUTPUTS  16
#define NUM_RULES    65536
#define BATCH        1024
#define MTILE        64                         // batch rows per workgroup (4 WMMA M-tiles)
#define MSUB         (MTILE / 16)               // 4 M sub-tiles per wave
#define KSPLIT       8                          // K split across blockIdx.y
#define KBLOCKS      (NUM_RULES / 32)           // 2048 K-steps of 32 rules
#define KB_PER_WG    (KBLOCKS / KSPLIT)         // 256 K-steps per workgroup
#define KUNROLL      4                          // K-steps per barrier (16 WMMAs / barrier)
#define NWAVES       9                          // one wave per i-index (9*16 = 144 = N)
#define BLOCK_THR    (NWAVES * 32)              // 288 threads, wave32
#define NTILES       (KBLOCKS * NWAVES)         // 18432 packed B tiles
#define BWS_BYTES    ((size_t)NTILES * 512 * 2) // 18.9 MB packed f16 B

__global__ void anfis_zero(float* __restrict__ out) {
    int i = blockIdx.x * blockDim.x + threadIdx.x;
    if (i < BATCH * NUM_OUTPUTS) out[i] = 0.0f;
}

// Pre-pack rule_params (f32, (65536,9,16)) into f16 WMMA B-fragment order:
// Bws[blk*9+j][lane][e], lane: n=lane&15, kh=lane>>4; e -> K = kh*8 + (e&7) + 16*(e>>3)
__global__ __launch_bounds__(256) void anfis_packB(const float* __restrict__ P,
                                                   _Float16* __restrict__ Bws) {
    const int wg   = blockIdx.x * 8 + (threadIdx.x >> 5);
    const int lane = threadIdx.x & 31;
    if (wg >= NTILES) return;
    const int blk = wg / 9, j = wg - blk * 9;
    const int n = lane & 15, kh = lane >> 4;
    const float* src = P + (size_t)(blk * 32 + kh * 8) * 144 + j * 16 + n;
    v16h hv;
    #pragma unroll
    for (int e = 0; e < 16; ++e) {
        const int koff = (e & 7) + ((e >> 3) << 4);
        hv[e] = (_Float16)src[(size_t)koff * 144];
    }
    *reinterpret_cast<v16h*>(Bws + (size_t)wg * 512 + lane * 16) = hv;
}

template <bool PRE>
__global__ __launch_bounds__(BLOCK_THR) void anfis_wmma(
    const float* __restrict__ x,        // (1024, 8)
    const float* __restrict__ centers,  // (8, 4)
    const float* __restrict__ widths,   // (8, 4)
    const float* __restrict__ P,        // (65536, 9, 16) f32 (fallback path)
    const _Float16* __restrict__ Bws,   // packed f16 B tiles (fast path)
    float* __restrict__ out)            // (1024, 16), pre-zeroed; atomic accumulate
{
    __shared__ float sh_mfs[MTILE][INPUT_DIM][NUM_MFS];
    __shared__ float sh_x[MTILE][INPUT_DIM];
    __shared__ float sh_inv[MTILE];
    __shared__ float sh_hiT[MTILE][256];  // prod of mfs digits 0..3 : index (r>>8)&255
    __shared__ float sh_loT[MTILE][256];  // prod of mfs digits 4..7 * invS : index r&255
    __shared__ __align__(32) _Float16 shA[2][KUNROLL][MSUB][32][16];

    const int tid   = threadIdx.x;
    const int mbase = blockIdx.x * MTILE;
    const int kb0   = blockIdx.y * KB_PER_WG;

    // membership functions (64 rows x 8 dims)
    for (int q = tid; q < MTILE * INPUT_DIM; q += BLOCK_THR) {
        const int b = q >> 3, i = q & 7;
        const float xv = x[(mbase + b) * INPUT_DIM + i];
        sh_x[b][i] = xv;
        #pragma unroll
        for (int m = 0; m < NUM_MFS; ++m) {
            const float c = centers[i * NUM_MFS + m];
            const float w = widths[i * NUM_MFS + m];
            const float d = xv - c;
            sh_mfs[b][i][m] = __expf(-(d * d) / (2.0f * w * w));
        }
    }
    __syncthreads();

    // normalizer factorizes: S[b] = prod_i (sum_m mfs[b,i,m])
    if (tid < MTILE) {
        float s = 1.0f;
        #pragma unroll
        for (int i = 0; i < INPUT_DIM; ++i)
            s *= sh_mfs[tid][i][0] + sh_mfs[tid][i][1] + sh_mfs[tid][i][2] + sh_mfs[tid][i][3];
        sh_inv[tid] = 1.0f / s;
    }
    __syncthreads();

    // two-level product tables: w[b,r] = hiT[b][(r>>8)&255] * loT[b][r&255]
    for (int q = tid; q < MTILE * 256; q += BLOCK_THR) {
        const int b = q >> 8, s = q & 255;
        const float m0 = sh_mfs[b][0][(s >> 6) & 3] * sh_mfs[b][1][(s >> 4) & 3];
        const float m1 = sh_mfs[b][2][(s >> 2) & 3] * sh_mfs[b][3][s & 3];
        sh_hiT[b][s] = m0 * m1;
        const float m2 = sh_mfs[b][4][(s >> 6) & 3] * sh_mfs[b][5][(s >> 4) & 3];
        const float m3 = sh_mfs[b][6][(s >> 2) & 3] * sh_mfs[b][7][s & 3];
        sh_loT[b][s] = m2 * m3 * sh_inv[b];
    }
    __syncthreads();

    // consumer setup
    const int wave = tid >> 5;          // i-index 0..8
    const int lane = tid & 31;
    const int ncol = lane & 15;
    const int kh   = lane >> 4;
    const _Float16* pB = Bws + ((size_t)kb0 * 9 + wave) * 512 + lane * 16;
    const float*   pBf = P + (size_t)(kb0 * 32 + kh * 8) * 144 + wave * 16 + ncol;

    // producer setup (threads 0..255): thread -> (row pb, 8 consecutive rules t0..t0+7)
    const int pb    = tid >> 2;                      // local batch row 0..63
    const int t0    = (tid & 3) * 8;                 // 0,8,16,24
    const int plane = (pb & 15) + (((t0 >> 3) & 1) << 4);
    const int pmt   = pb >> 4;                       // which 16-row M tile (0..3)
    const int pe0   = (t0 & 7) + ((t0 >> 4) << 3);   // 0 or 8; 8 contiguous e's

    v8f c[MSUB] = {};
    const int OUTER = KB_PER_WG / KUNROLL;

    for (int it = 0; it < OUTER; ++it) {
        const int base = kb0 + it * KUNROLL;
        const int buf  = it & 1;

        if (tid < 256) {
            const float hi = sh_hiT[pb][(base >> 3) & 255];  // constant across the 4 sub-blocks
            #pragma unroll
            for (int u = 0; u < KUNROLL; ++u) {
                const int lo = (((base & 7) + u) << 5) | t0; // low 8 bits of rule index
                const v8f q8 = *reinterpret_cast<const v8f*>(&sh_loT[pb][lo]);
                v8h hv;
                #pragma unroll
                for (int j = 0; j < 8; ++j)
                    hv[j] = (_Float16)(hi * q8[j]);
                *reinterpret_cast<v8h*>(&shA[buf][u][pmt][plane][pe0]) = hv; // ds_store_b128
            }
        }
        __syncthreads();  // single barrier; double-buffer + pre-WMMA dscnt waits cover WAR

        #pragma unroll
        for (int u = 0; u < KUNROLL; ++u) {
            v16h bfrag;
            if (PRE) {
                bfrag = *reinterpret_cast<const v16h*>(pB + (size_t)u * 9 * 512);
            } else {
                #pragma unroll
                for (int e = 0; e < 16; ++e) {
                    const int koff = u * 32 + (e & 7) + ((e >> 3) << 4);
                    bfrag[e] = (_Float16)pBf[(size_t)koff * 144];
                }
            }
            #pragma unroll
            for (int mt = 0; mt < MSUB; ++mt) {
                const v16h a = *reinterpret_cast<const v16h*>(&shA[buf][u][mt][lane][0]);
                c[mt] = __builtin_amdgcn_wmma_f32_16x16x32_f16(false, a, false, bfrag,
                                                               (short)0, c[mt], false, false);
            }
        }
        if (PRE) pB  += (size_t)KUNROLL * 9 * 512;
        else     pBf += (size_t)KUNROLL * 32 * 144;
    }

    // epilogue: out[b,n] += sum_i C_i[b,n] * xb[b,i]   (wave = i; wave 8 -> xb = 1)
    #pragma unroll
    for (int mt = 0; mt < MSUB; ++mt) {
        #pragma unroll
        for (int v = 0; v < 8; ++v) {
            const int lb = mt * 16 + kh * 8 + v;  // C layout: lanes 0-15 M=v, lanes 16-31 M=v+8
            const float xbv = (wave < 8) ? sh_x[lb][wave] : 1.0f;
            __hip_atomic_fetch_add(&out[(mbase + lb) * 16 + ncol], c[mt][v] * xbv,
                                   __ATOMIC_RELAXED, __HIP_MEMORY_SCOPE_AGENT);
        }
    }
}

extern "C" void kernel_launch(void* const* d_in, const int* in_sizes, int n_in,
                              void* d_out, int out_size, void* d_ws, size_t ws_size,
                              hipStream_t stream) {
    (void)in_sizes; (void)n_in; (void)out_size;
    const float* x       = (const float*)d_in[0];
    const float* centers = (const float*)d_in[1];
    const float* widths  = (const float*)d_in[2];
    const float* P       = (const float*)d_in[3];
    float* out           = (float*)d_out;

    const bool pre = (d_ws != nullptr) && (ws_size >= BWS_BYTES);
    anfis_zero<<<(BATCH * NUM_OUTPUTS + 255) / 256, 256, 0, stream>>>(out);
    if (pre) {
        _Float16* Bws = (_Float16*)d_ws;
        anfis_packB<<<(NTILES + 7) / 8, 256, 0, stream>>>(P, Bws);
        anfis_wmma<true><<<dim3(BATCH / MTILE, KSPLIT), BLOCK_THR, 0, stream>>>(
            x, centers, widths, P, Bws, out);
    } else {
        anfis_wmma<false><<<dim3(BATCH / MTILE, KSPLIT), BLOCK_THR, 0, stream>>>(
            x, centers, widths, P, nullptr, out);
    }
}
